// MultiHeadAttention_22033182228520
// MI455X (gfx1250) — matrix operand order repaired and checked by reading, not compiled, not run on hardware
//
#include <hip/hip_runtime.h>
#include <hip/hip_bf16.h>

typedef __attribute__((ext_vector_type(16))) _Float16 v16h;
typedef __attribute__((ext_vector_type(8)))  _Float16 v8h;
typedef __attribute__((ext_vector_type(8)))  float    v8f;

#define D_MODEL 512
#define L_SEQ   4096
#define N_HEADS 8
#define D_HEAD  64

// ---------------------------------------------------------------------------
// WMMA fragment helpers (wave32, v_wmma_f32_16x16x32_f16)
// A (16x32 f16): lane r=lane&15 holds row r; half=lane>>4 selects K-groups
//   v0..7 = K[half*16 .. half*16+15]
// B (32x16 f16): lane n=lane&15 is the output column; kb=(lane>>4)*16;
//   v0..7 = K[kb .. kb+15] -> one 32B contiguous load when B is stored
//   column-contiguous ("Bt[n][k]").
// C/D (16x16 f32): reg i, lanes 0-15 -> (M=i, N=lane); lanes 16-31 -> (M=8+i).
// ---------------------------------------------------------------------------
static __device__ __forceinline__ v16h load_a_frag(const _Float16* p, int ld, int lane) {
    int r = lane & 15, hf = lane >> 4;
    const _Float16* row = p + r * ld + hf * 16;
    v8h lo = *(const v8h*)(row);
    v8h hi = *(const v8h*)(row + 8);
    v16h a;
#pragma unroll
    for (int i = 0; i < 8; ++i) { a[i] = lo[i]; a[i + 8] = hi[i]; }
    return a;
}

static __device__ __forceinline__ v16h load_b_frag(const _Float16* p, int ld, int lane) {
    int n = lane & 15, kb = (lane >> 4) * 16;
    return *(const v16h*)(p + (size_t)n * ld + kb);   // 32 contiguous bytes
}

static __device__ __forceinline__ v8f wmma_f16(v16h a, v16h b, v8f c) {
    return __builtin_amdgcn_wmma_f32_16x16x32_f16(false, a, false, b, (short)0, c,
                                                  false, false);
}

// ---------------------------------------------------------------------------
// Conversion kernels
// ---------------------------------------------------------------------------
__global__ void k_cvt_f16(const float* __restrict__ x, _Float16* __restrict__ y, int n) {
    int i = blockIdx.x * blockDim.x + threadIdx.x;
    if (i < n) y[i] = (_Float16)x[i];
}

// Wt[n][k] = W[k][n]  (column-contiguous f16 copy of a 512x512 weight)
__global__ void k_cvt_wt(const float* __restrict__ W, _Float16* __restrict__ Wt) {
    int i = blockIdx.x * blockDim.x + threadIdx.x;   // 0 .. 512*512-1
    int k = i >> 9, n = i & 511;
    Wt[n * D_MODEL + k] = (_Float16)W[k * D_MODEL + n];
}

// ---------------------------------------------------------------------------
// GEMM: Y = Ah(f16,8192x512) @ W(f16,col-contig) + b. One wave -> 32x64 tile.
// Parity-indexed double buffer in a fully unrolled K loop; launch_bounds
// lifts the VGPR cap so nothing spills (~180 VGPRs/wave, 2 waves/SIMD).
// vmode 0: f16 row-major store (Q,K)   vmode 1: V transposed per head
// vmode 2: f32 row-major store (final output)
// ---------------------------------------------------------------------------
__global__ void __launch_bounds__(256, 1)
k_gemm(const _Float16* __restrict__ Ah, const _Float16* __restrict__ Wt,
       const float* __restrict__ bias, void* __restrict__ Yv, int vmode) {
    int lane = threadIdx.x & 31, wave = threadIdx.x >> 5;
    int task  = blockIdx.x * 8 + wave;        // 256 mtiles * 8 ntiles = 2048 tasks
    int mtile = (task >> 3) * 32;
    int ntile = (task & 7) * 64;

    const _Float16* a0 = Ah + (size_t)mtile * D_MODEL;
    const _Float16* a1 = a0 + 16 * D_MODEL;
    const _Float16* wb = Wt + (size_t)ntile * D_MODEL;

    v8f acc[2][4] = {};
    v16h af[2][2], bf[2][4];
    af[0][0] = load_a_frag(a0, D_MODEL, lane);
    af[0][1] = load_a_frag(a1, D_MODEL, lane);
#pragma unroll
    for (int j = 0; j < 4; ++j)
        bf[0][j] = load_b_frag(wb + (size_t)j * 16 * D_MODEL, D_MODEL, lane);

#pragma unroll
    for (int s = 0; s < 15; ++s) {            // k = 32*(s+1) loads, k = 32*s math
        int cur = s & 1, nxt = cur ^ 1;
        int k = (s + 1) * 32;
        af[nxt][0] = load_a_frag(a0 + k, D_MODEL, lane);
        af[nxt][1] = load_a_frag(a1 + k, D_MODEL, lane);
#pragma unroll
        for (int j = 0; j < 4; ++j)
            bf[nxt][j] = load_b_frag(wb + (size_t)j * 16 * D_MODEL + k, D_MODEL, lane);
#pragma unroll
        for (int j = 0; j < 4; ++j) {
            acc[0][j] = wmma_f16(af[cur][0], bf[cur][j], acc[0][j]);
            acc[1][j] = wmma_f16(af[cur][1], bf[cur][j], acc[1][j]);
        }
    }
#pragma unroll
    for (int j = 0; j < 4; ++j) {             // tail: buffer parity 1 (s==15)
        acc[0][j] = wmma_f16(af[1][0], bf[1][j], acc[0][j]);
        acc[1][j] = wmma_f16(af[1][1], bf[1][j], acc[1][j]);
    }

    int hf = lane >> 4, nc = lane & 15;
#pragma unroll
    for (int j = 0; j < 4; ++j) {
        float bj = bias[ntile + j * 16 + nc];
#pragma unroll
        for (int i = 0; i < 8; ++i) { acc[0][j][i] += bj; acc[1][j][i] += bj; }
    }

    if (vmode == 0) {
        _Float16* Y = (_Float16*)Yv;
#pragma unroll
        for (int t = 0; t < 2; ++t)
#pragma unroll
            for (int j = 0; j < 4; ++j)
#pragma unroll
                for (int i = 0; i < 8; ++i)
                    Y[(size_t)(mtile + t * 16 + hf * 8 + i) * D_MODEL + ntile + j * 16 + nc] =
                        (_Float16)acc[t][j][i];
    } else if (vmode == 1) {
        _Float16* Y = (_Float16*)Yv;
#pragma unroll
        for (int t = 0; t < 2; ++t) {
            int m0   = mtile + t * 16 + hf * 8;
            int bidx = m0 >> 12;               // /4096
            int l0   = m0 & (L_SEQ - 1);
#pragma unroll
            for (int j = 0; j < 4; ++j) {
                int n = ntile + j * 16 + nc;
                int h = n >> 6, d = n & 63;
                v8h ph;
#pragma unroll
                for (int i = 0; i < 8; ++i) ph[i] = (_Float16)acc[t][j][i];
                *(v8h*)(Y + ((size_t)((bidx * N_HEADS + h) * D_HEAD + d)) * L_SEQ + l0) = ph;
            }
        }
    } else {
        float* Y = (float*)Yv;
#pragma unroll
        for (int t = 0; t < 2; ++t)
#pragma unroll
            for (int j = 0; j < 4; ++j)
#pragma unroll
                for (int i = 0; i < 8; ++i)
                    Y[(size_t)(mtile + t * 16 + hf * 8 + i) * D_MODEL + ntile + j * 16 + nc] =
                        acc[t][j][i];
    }
}

// ---------------------------------------------------------------------------
// Flash attention: one wave owns a 16-row query tile of one (batch, head).
// 64 keys per block: 8 score WMMAs, one softmax pass, 8 PV WMMAs.
// V fragments are issued before the softmax VALU chain to hide their latency;
// the next K slab is prefetched (global_prefetch_b8) with per-lane rows.
// ---------------------------------------------------------------------------
__global__ void __launch_bounds__(256, 1)
k_attn(const _Float16* __restrict__ Qh, const _Float16* __restrict__ Kh,
       const _Float16* __restrict__ Vt, const unsigned char* __restrict__ mask,
       _Float16* __restrict__ Ah) {
    __shared__ __align__(32) _Float16 plds[8][16 * 64];   // per-wave P tile, 16 KB total

    int lane = threadIdx.x & 31, wave = threadIdx.x >> 5;
    int task = blockIdx.x * 8 + wave;          // 2*8*256 = 4096 tasks
    int mt   = task & 255;
    int bh   = task >> 8;
    int b = bh >> 3, h = bh & 7;

    const _Float16* qbase = Qh + (size_t)(b * L_SEQ + mt * 16) * D_MODEL + h * D_HEAD;
    const _Float16* kbase = Kh + (size_t)b * L_SEQ * D_MODEL + h * D_HEAD;
    const _Float16* vbase = Vt + (size_t)bh * D_HEAD * L_SEQ;
    const unsigned char* mbase = mask + (size_t)b * L_SEQ;

    v16h qf0 = load_a_frag(qbase, D_MODEL, lane);        // DH 0..31
    v16h qf1 = load_a_frag(qbase + 32, D_MODEL, lane);   // DH 32..63

    float mrow[8], lrow[8];
    v8f o[4] = {};
#pragma unroll
    for (int i = 0; i < 8; ++i) { mrow[i] = -3.0e38f; lrow[i] = 0.0f; }

    const float scale = 0.125f;                // 1/sqrt(64)
    const float L2E   = 1.44269504f;
    int nc = lane & 15, hf = lane >> 4;

#pragma unroll 1
    for (int n0 = 0; n0 < L_SEQ; n0 += 64) {
        // ---- all K fragments for 64 keys, then 8 score WMMAs ----
        v16h kb[8];
#pragma unroll
        for (int t = 0; t < 4; ++t) {
            kb[t]     = load_b_frag(kbase + (size_t)(n0 + t * 16) * D_MODEL, D_MODEL, lane);
            kb[4 + t] = load_b_frag(kbase + (size_t)(n0 + t * 16) * D_MODEL + 32, D_MODEL, lane);
        }
        v8f s[4] = {};
#pragma unroll
        for (int t = 0; t < 4; ++t) {
            s[t] = wmma_f16(qf0, kb[t], s[t]);
            s[t] = wmma_f16(qf1, kb[4 + t], s[t]);
        }

        // ---- V fragments in flight during the softmax VALU chain ----
        v16h vb[8];
#pragma unroll
        for (int j = 0; j < 4; ++j) {
            vb[j]     = load_b_frag(vbase + (size_t)(j * 16) * L_SEQ + n0, L_SEQ, lane);
            vb[4 + j] = load_b_frag(vbase + (size_t)(j * 16) * L_SEQ + n0 + 32, L_SEQ, lane);
        }

        // ---- prefetch next 64-row K slab (global_prefetch_b8) ----
        if (n0 + 64 < L_SEQ) {
            __builtin_prefetch(kbase + (size_t)(n0 + 64 + lane) * D_MODEL, 0, 3);
            __builtin_prefetch(kbase + (size_t)(n0 + 96 + lane) * D_MODEL, 0, 3);
        }

        float mb[4];
#pragma unroll
        for (int t = 0; t < 4; ++t)
            mb[t] = mbase[n0 + t * 16 + nc] ? -1.0e30f : 0.0f;

        // ---- online softmax; one reduction tree per row for all 64 keys ----
#pragma unroll
        for (int i = 0; i < 8; ++i) {
            float a[4];
#pragma unroll
            for (int t = 0; t < 4; ++t) a[t] = s[t][i] * scale + mb[t];
            float v = fmaxf(fmaxf(a[0], a[1]), fmaxf(a[2], a[3]));
#pragma unroll
            for (int off = 1; off < 16; off <<= 1)
                v = fmaxf(v, __shfl_xor(v, off, 16));
            float mnew = fmaxf(mrow[i], v);
            float resc = __builtin_exp2f((mrow[i] - mnew) * L2E);
            mrow[i] = mnew;
            float p[4], ps = 0.0f;
#pragma unroll
            for (int t = 0; t < 4; ++t) { p[t] = __builtin_exp2f((a[t] - mnew) * L2E); ps += p[t]; }
#pragma unroll
            for (int off = 1; off < 16; off <<= 1)
                ps += __shfl_xor(ps, off, 16);
            lrow[i] = lrow[i] * resc + ps;
#pragma unroll
            for (int j = 0; j < 4; ++j) o[j][i] *= resc;
            int m = hf * 8 + i;
#pragma unroll
            for (int t = 0; t < 4; ++t)
                plds[wave][m * 64 + t * 16 + nc] = (_Float16)p[t];
        }

        // ---- O += P * V ----
        v16h pa0 = load_a_frag(&plds[wave][0], 64, lane);    // keys n0..n0+31
        v16h pa1 = load_a_frag(&plds[wave][32], 64, lane);   // keys n0+32..n0+63
#pragma unroll
        for (int j = 0; j < 4; ++j) {
            o[j] = wmma_f16(pa0, vb[j], o[j]);
            o[j] = wmma_f16(pa1, vb[4 + j], o[j]);
        }
    }

    // ---- normalize and store f16 row-major attention output ----
    size_t obase = (size_t)(b * L_SEQ + mt * 16) * D_MODEL + h * D_HEAD;
#pragma unroll
    for (int i = 0; i < 8; ++i) {
        float inv = 1.0f / lrow[i];
#pragma unroll
        for (int j = 0; j < 4; ++j)
            Ah[obase + (size_t)(hf * 8 + i) * D_MODEL + j * 16 + nc] =
                (_Float16)(o[j][i] * inv);
    }
}

// ---------------------------------------------------------------------------
// Host launcher
// ---------------------------------------------------------------------------
extern "C" void kernel_launch(void* const* d_in, const int* in_sizes, int n_in,
                              void* d_out, int out_size, void* d_ws, size_t ws_size,
                              hipStream_t stream) {
    const float* x  = (const float*)d_in[0];
    const unsigned char* mask = (const unsigned char*)d_in[1];
    const float* Wq = (const float*)d_in[2];
    const float* bq = (const float*)d_in[3];
    const float* Wk = (const float*)d_in[4];
    const float* bk = (const float*)d_in[5];
    const float* Wv = (const float*)d_in[6];
    const float* bv = (const float*)d_in[7];
    const float* Wo = (const float*)d_in[8];
    const float* bo = (const float*)d_in[9];

    char* ws = (char*)d_ws;
    const size_t MB8 = 8u * 1024u * 1024u;
    _Float16* xh  = (_Float16*)(ws + 0 * MB8);
    _Float16* Qh  = (_Float16*)(ws + 1 * MB8);
    _Float16* Kh  = (_Float16*)(ws + 2 * MB8);
    _Float16* Vth = (_Float16*)(ws + 3 * MB8);
    _Float16* Ahb = (_Float16*)(ws + 4 * MB8);
    _Float16* Wqt = (_Float16*)(ws + 5 * MB8);
    _Float16* Wkt = (_Float16*)(ws + 5 * MB8 + 524288);
    _Float16* Wvt = (_Float16*)(ws + 5 * MB8 + 2 * 524288);
    _Float16* Wot = (_Float16*)(ws + 5 * MB8 + 3 * 524288);

    const int NX = 2 * L_SEQ * D_MODEL;           // 4,194,304 elements
    k_cvt_f16<<<NX / 256, 256, 0, stream>>>(x, xh, NX);
    k_cvt_wt<<<(D_MODEL * D_MODEL) / 256, 256, 0, stream>>>(Wq, Wqt);
    k_cvt_wt<<<(D_MODEL * D_MODEL) / 256, 256, 0, stream>>>(Wk, Wkt);
    k_cvt_wt<<<(D_MODEL * D_MODEL) / 256, 256, 0, stream>>>(Wv, Wvt);
    k_cvt_wt<<<(D_MODEL * D_MODEL) / 256, 256, 0, stream>>>(Wo, Wot);

    // 256 M-tiles(32) * 8 N-tiles(64) = 2048 wave tasks, 8 waves/block
    k_gemm<<<256, 256, 0, stream>>>(xh, Wqt, bq, Qh, 0);
    k_gemm<<<256, 256, 0, stream>>>(xh, Wkt, bk, Kh, 0);
    k_gemm<<<256, 256, 0, stream>>>(xh, Wvt, bv, Vth, 1);

    // 16 (b,h) * 256 query tiles = 4096 wave tasks -> 512 blocks
    k_attn<<<512, 256, 0, stream>>>(Qh, Kh, Vth, mask, Ahb);

    k_gemm<<<256, 256, 0, stream>>>(Ahb, Wot, bo, (void*)d_out, 2);
}
